// GNNLayer_48619029791192
// MI455X (gfx1250) — matrix operand order repaired and checked by reading, not compiled, not run on hardware
//
#include <hip/hip_runtime.h>
#include <stdint.h>

typedef __attribute__((ext_vector_type(16))) __bf16 v16bf;
typedef __attribute__((ext_vector_type(8)))  float  v8f;
typedef __attribute__((ext_vector_type(4)))  unsigned int u32x4;
typedef __attribute__((ext_vector_type(2)))  unsigned int u32x2;
typedef __attribute__((ext_vector_type(4)))  float  f32x4;

#define D      128
#define NT     8          // 8 column tiles of 16 -> 128 outputs
#define KT_TR  5          // padded K: 160 / 32
#define KT_G2  4          // 128 / 32
#define KPAD   160
#define TK     4          // number of keys
#define TT     2          // hops per key
#define KROWS_TR 131      // D_IN + OHL
#define NB     128        // blocks per key in the edge kernel

// packed weight region sizes (in ushorts)
#define PW_LIN  (KT_G2 * NT * 32 * 16)   // 16384
#define PW_TR1  (KT_TR * NT * 32 * 16)   // 20480 per key
#define PW_G21  (KT_G2 * NT * 32 * 16)   // 16384 per key

// LDS layout for edge kernel (ushort offsets)
#define LDS_WG1  0
#define LDS_WG2  (PW_TR1)
#define LDS_WTR  (PW_TR1 + PW_G21)
#define LDS_HT   (2 * PW_TR1 + PW_G21)            // 4 waves x 16*KPAD
#define LDS_UT   (LDS_HT + 4 * 16 * KPAD)         // 4 waves x 16*D
#define LDS_TOTAL (LDS_UT + 4 * 16 * D)           // 75776 ushorts = 151552 B

union Frag { v16bf bf; u32x4 q[2]; };

__device__ __forceinline__ unsigned f2bf1(float f) {
  unsigned u = __float_as_uint(f);
  return (u + 0x7FFFu + ((u >> 16) & 1u)) >> 16;   // round-to-nearest-even bf16
}
__device__ __forceinline__ unsigned f2bf2(float lo, float hi) {
  return f2bf1(lo) | (f2bf1(hi) << 16);
}
__device__ __forceinline__ float sigm(float x) {
  return 1.0f / (1.0f + __expf(-x));
}
__device__ __forceinline__ v8f vzero8() {
  v8f z;
#pragma unroll
  for (int j = 0; j < 8; ++j) z[j] = 0.0f;
  return z;
}

// ---------------------------------------------------------------------------
// Repack fp32 weights into bf16 WMMA B-fragment layout:
//   packed[((kt*8 + nt)*32 + lane)*16 + e] = W[kt*32 + (e/8)*16 + (lane/16)*8 + e%8][nt*16 + lane%16]
// Zero-pad K beyond the true row count (131 -> 160 for W_tr / W_g1).
// ---------------------------------------------------------------------------
__global__ void convert_weights(const float* __restrict__ Wlin,
                                const float* __restrict__ Wtr,
                                const float* __restrict__ Wg1,
                                const float* __restrict__ Wg2,
                                unsigned short* __restrict__ ws, int total) {
  int idx = blockIdx.x * blockDim.x + threadIdx.x;
  if (idx >= total) return;

  const float* src;
  unsigned short* dst;
  int local, krows;

  if (idx < PW_LIN) {
    local = idx; src = Wlin; krows = D; dst = ws;
  } else if (idx < PW_LIN + TK * PW_TR1) {
    int t = idx - PW_LIN;
    int kk = t / PW_TR1; local = t - kk * PW_TR1;
    src = Wtr + (size_t)kk * KROWS_TR * D; krows = KROWS_TR;
    dst = ws + PW_LIN + (size_t)kk * PW_TR1;
  } else if (idx < PW_LIN + 2 * TK * PW_TR1) {
    int t = idx - (PW_LIN + TK * PW_TR1);
    int kk = t / PW_TR1; local = t - kk * PW_TR1;
    src = Wg1 + (size_t)kk * KROWS_TR * D; krows = KROWS_TR;
    dst = ws + PW_LIN + (size_t)TK * PW_TR1 + (size_t)kk * PW_TR1;
  } else {
    int t = idx - (PW_LIN + 2 * TK * PW_TR1);
    int kk = t / PW_G21; local = t - kk * PW_G21;
    src = Wg2 + (size_t)kk * D * D; krows = D;
    dst = ws + PW_LIN + (size_t)2 * TK * PW_TR1 + (size_t)kk * PW_G21;
  }

  int e    = local & 15;
  int lane = (local >> 4) & 31;
  int nt   = (local >> 9) & 7;
  int kt   = local >> 12;
  int K    = kt * 32 + (e >> 3) * 16 + (lane >> 4) * 8 + (e & 7);
  int Ncol = nt * 16 + (lane & 15);
  float v  = (K < krows) ? src[(size_t)K * D + Ncol] : 0.0f;
  dst[local] = (unsigned short)f2bf1(v);
}

// ---------------------------------------------------------------------------
// h3 = h @ W_lin + b_lin   (also serves as the output initializer)
// One wave handles 16 rows; 4 waves per block, private LDS tile per wave.
// ---------------------------------------------------------------------------
__global__ void linear_kernel(const float* __restrict__ h,
                              const unsigned short* __restrict__ pW,
                              const float* __restrict__ bias,
                              float* __restrict__ out, int nRows) {
  __shared__ unsigned short sh[4][16 * D];
  int wid = threadIdx.x >> 5, lane = threadIdx.x & 31;
  int tile = blockIdx.x * 4 + wid;
  int row0 = tile * 16;
  if (row0 >= nRows) return;
  int half = lane >> 4, lm = lane & 15;
  unsigned short* ht = sh[wid];

  for (int r = 0; r < 16; ++r) {
    f32x4 v = *((const f32x4*)(h + (size_t)(row0 + r) * D) + lane);
    u32x2 p = { f2bf2(v.x, v.y), f2bf2(v.z, v.w) };
    *(u32x2*)(ht + r * D + lane * 4) = p;
  }
  asm volatile("s_wait_dscnt 0" ::: "memory");

  v8f acc[NT];
#pragma unroll
  for (int nt = 0; nt < NT; ++nt) acc[nt] = vzero8();

#pragma unroll
  for (int kt = 0; kt < KT_G2; ++kt) {
    Frag a;
    const unsigned short* ar = ht + lm * D + kt * 32 + half * 8;
    a.q[0] = *(const u32x4*)ar;
    a.q[1] = *(const u32x4*)(ar + 16);
#pragma unroll
    for (int nt = 0; nt < NT; ++nt) {
      Frag b;
      const u32x4* bp = (const u32x4*)(pW + (size_t)((kt * NT + nt) * 32 + lane) * 16);
      b.q[0] = bp[0]; b.q[1] = bp[1];
      acc[nt] = __builtin_amdgcn_wmma_f32_16x16x32_bf16(
          false, a.bf, false, b.bf, (short)0, acc[nt], false, false);
    }
  }

#pragma unroll
  for (int nt = 0; nt < NT; ++nt) {
    float bv = bias[nt * 16 + lm];
#pragma unroll
    for (int j = 0; j < 8; ++j)
      out[(size_t)(row0 + j + half * 8) * D + nt * 16 + lm] = acc[nt][j] + bv;
  }
}

// ---------------------------------------------------------------------------
// Per-key edge pipeline, persistent blocks:
//   - stage the key's packed bf16 weights (112 KB) in LDS once per block
//   - 4 waves grid-stride over 16-edge tiles
//   - per tile/hop: gather -> 3 bf16 WMMA GEMMs (A and B from LDS) ->
//     gated product -> scaled f32 atomic scatter-add
// ---------------------------------------------------------------------------
__global__ void edge_kernel(const float* __restrict__ h,
                            const int* __restrict__ pairs,
                            const float* __restrict__ degrees,
                            const int* __restrict__ scatter,
                            const unsigned short* __restrict__ pWtr,
                            const unsigned short* __restrict__ pWg1,
                            const unsigned short* __restrict__ pWg2,
                            const float* __restrict__ b_tr,
                            const float* __restrict__ b_g1,
                            const float* __restrict__ b_g2,
                            const float* __restrict__ eps,
                            float* __restrict__ out, int E) {
  extern __shared__ unsigned short smem[];
  const int tid = threadIdx.x;
  const int wid = tid >> 5, lane = tid & 31;
  const int k = blockIdx.x / NB;
  const int b = blockIdx.x - k * NB;
  const int half = lane >> 4, lm = lane & 15;

  unsigned short* lwg1 = smem + LDS_WG1;
  unsigned short* lwg2 = smem + LDS_WG2;
  unsigned short* lwtr = smem + LDS_WTR;
  unsigned short* ht   = smem + LDS_HT + wid * (16 * KPAD);
  unsigned short* ut   = smem + LDS_UT + wid * (16 * D);

  // ---- stage this key's packed weights into LDS (cooperative, then barrier)
  {
    const unsigned short* g1 = pWg1 + (size_t)k * PW_TR1;
    const unsigned short* g2 = pWg2 + (size_t)k * PW_G21;
    const unsigned short* tr = pWtr + (size_t)k * PW_TR1;
    const int totalVec = (2 * PW_TR1 + PW_G21) / 8;   // 16-byte vectors
    for (int v = tid; v < totalVec; v += 128) {
      int off = v * 8;
      const unsigned short* s;
      if (off < PW_TR1)               s = g1 + off;
      else if (off < PW_TR1 + PW_G21) s = g2 + (off - PW_TR1);
      else                            s = tr + (off - PW_TR1 - PW_G21);
      *(u32x4*)(smem + off) = *(const u32x4*)s;
    }
  }
  __syncthreads();

  // ---- per-key constants hoisted to registers
  float bg1v[NT], bg2v[NT], btv[NT];
#pragma unroll
  for (int nt = 0; nt < NT; ++nt) {
    bg1v[nt] = b_g1[k * D + nt * 16 + lm];
    bg2v[nt] = b_g2[k * D + nt * 16 + lm];
    btv[nt]  = b_tr[k * D + nt * 16 + lm];
  }
  const float scale = 1.0f + eps[k];
  const int tiles16 = E >> 4;

  for (int wt = b * 4 + wid; wt < tiles16; wt += NB * 4) {
    const int e0 = wt * 16;

    v8f hp[NT];
#pragma unroll
    for (int nt = 0; nt < NT; ++nt)
#pragma unroll
      for (int j = 0; j < 8; ++j) hp[nt][j] = 1.0f;

#pragma unroll
    for (int i = 0; i < TT; ++i) {
      asm volatile("s_wait_dscnt 0" ::: "memory");  // WAR vs previous reads of ht/ut
      // ---- stage gathered ht tile: cols [0,128)=h[pairs], [128,131)=degrees, rest 0
      const int* prow = pairs + ((size_t)(k * TT + i)) * E + e0;
      for (int r = 0; r < 16; ++r) {
        int node = prow[r];
        f32x4 v = *((const f32x4*)(h + (size_t)node * D) + lane);
        u32x2 p = { f2bf2(v.x, v.y), f2bf2(v.z, v.w) };
        *(u32x2*)(ht + r * KPAD + lane * 4) = p;
      }
      {
        u32x4 z = {0, 0, 0, 0};
        u32x4 d = z;
        if (half == 0) {
          const float* dg = degrees + (((size_t)(k * TT + i)) * E + (e0 + lm)) * 3;
          d.x = f2bf2(dg[0], dg[1]);
          d.y = f2bf1(dg[2]);
        }
        *(u32x4*)(ht + lm * KPAD + 128 + half * 16) = d;
        *(u32x4*)(ht + lm * KPAD + 136 + half * 16) = z;
      }
      asm volatile("s_wait_dscnt 0" ::: "memory");

      // ---- GEMM: u_pre = ht @ W_g1   (A and B fragments from LDS)
      v8f acc[NT];
#pragma unroll
      for (int nt = 0; nt < NT; ++nt) acc[nt] = vzero8();
#pragma unroll
      for (int kt = 0; kt < KT_TR; ++kt) {
        Frag a;
        const unsigned short* ar = ht + lm * KPAD + kt * 32 + half * 8;
        a.q[0] = *(const u32x4*)ar;
        a.q[1] = *(const u32x4*)(ar + 16);
#pragma unroll
        for (int nt = 0; nt < NT; ++nt) {
          Frag bb;
          const u32x4* bp = (const u32x4*)(lwg1 + (size_t)((kt * NT + nt) * 32 + lane) * 16);
          bb.q[0] = bp[0]; bb.q[1] = bp[1];
          acc[nt] = __builtin_amdgcn_wmma_f32_16x16x32_bf16(
              false, a.bf, false, bb.bf, (short)0, acc[nt], false, false);
        }
      }
      // relu(+bias) -> LDS bf16 u tile
#pragma unroll
      for (int nt = 0; nt < NT; ++nt) {
#pragma unroll
        for (int j = 0; j < 8; ++j) {
          float u = acc[nt][j] + bg1v[nt];
          u = u > 0.0f ? u : 0.0f;
          ut[(j + half * 8) * D + nt * 16 + lm] = (unsigned short)f2bf1(u);
        }
      }
      asm volatile("s_wait_dscnt 0" ::: "memory");

      // ---- GEMM: g_pre = u @ W_g2, then g = sigmoid(g_pre + b_g2)
      v8f accg[NT];
#pragma unroll
      for (int nt = 0; nt < NT; ++nt) accg[nt] = vzero8();
#pragma unroll
      for (int kt = 0; kt < KT_G2; ++kt) {
        Frag a;
        const unsigned short* ar = ut + lm * D + kt * 32 + half * 8;
        a.q[0] = *(const u32x4*)ar;
        a.q[1] = *(const u32x4*)(ar + 16);
#pragma unroll
        for (int nt = 0; nt < NT; ++nt) {
          Frag bb;
          const u32x4* bp = (const u32x4*)(lwg2 + (size_t)((kt * NT + nt) * 32 + lane) * 16);
          bb.q[0] = bp[0]; bb.q[1] = bp[1];
          accg[nt] = __builtin_amdgcn_wmma_f32_16x16x32_bf16(
              false, a.bf, false, bb.bf, (short)0, accg[nt], false, false);
        }
      }
#pragma unroll
      for (int nt = 0; nt < NT; ++nt)
#pragma unroll
        for (int j = 0; j < 8; ++j) accg[nt][j] = sigm(accg[nt][j] + bg2v[nt]);

      // ---- GEMM: tr_pre = ht @ W_tr; hp *= sigmoid(tr_pre + b_tr) * g
#pragma unroll
      for (int nt = 0; nt < NT; ++nt) acc[nt] = vzero8();
#pragma unroll
      for (int kt = 0; kt < KT_TR; ++kt) {
        Frag a;
        const unsigned short* ar = ht + lm * KPAD + kt * 32 + half * 8;
        a.q[0] = *(const u32x4*)ar;
        a.q[1] = *(const u32x4*)(ar + 16);
#pragma unroll
        for (int nt = 0; nt < NT; ++nt) {
          Frag bb;
          const u32x4* bp = (const u32x4*)(lwtr + (size_t)((kt * NT + nt) * 32 + lane) * 16);
          bb.q[0] = bp[0]; bb.q[1] = bp[1];
          acc[nt] = __builtin_amdgcn_wmma_f32_16x16x32_bf16(
              false, a.bf, false, bb.bf, (short)0, acc[nt], false, false);
        }
      }
#pragma unroll
      for (int nt = 0; nt < NT; ++nt)
#pragma unroll
        for (int j = 0; j < 8; ++j)
          hp[nt][j] *= sigm(acc[nt][j] + btv[nt]) * accg[nt][j];
    }

    // ---- scatter-add: out[scatter[k,e], :] += (1+eps_k) * hp
    int sv = (lane < 16) ? scatter[(size_t)k * E + e0 + lane] : 0;
#pragma unroll
    for (int nt = 0; nt < NT; ++nt) {
#pragma unroll
      for (int j = 0; j < 8; ++j) {
        int sidx = __shfl(sv, j + half * 8, 32);
        float val = hp[nt][j] * scale;
        __hip_atomic_fetch_add(out + (size_t)sidx * D + nt * 16 + lm, val,
                               __ATOMIC_RELAXED, __HIP_MEMORY_SCOPE_AGENT);
      }
    }
  }
}

extern "C" void kernel_launch(void* const* d_in, const int* in_sizes, int n_in,
                              void* d_out, int out_size, void* d_ws, size_t ws_size,
                              hipStream_t stream) {
  const float* h       = (const float*)d_in[0];
  const int*   pairs   = (const int*)d_in[1];
  const float* degrees = (const float*)d_in[2];
  const int*   scatter = (const int*)d_in[3];
  const float* W_lin   = (const float*)d_in[4];
  const float* b_lin   = (const float*)d_in[5];
  const float* W_tr    = (const float*)d_in[6];
  const float* b_tr    = (const float*)d_in[7];
  const float* W_g1    = (const float*)d_in[8];
  const float* b_g1    = (const float*)d_in[9];
  const float* W_g2    = (const float*)d_in[10];
  const float* b_g2    = (const float*)d_in[11];
  const float* eps     = (const float*)d_in[12];

  int N = in_sizes[0] / D;        // 50000
  int E = in_sizes[3] / TK;       // 100000

  unsigned short* ws    = (unsigned short*)d_ws;
  unsigned short* pWlin = ws;
  unsigned short* pWtr  = ws + PW_LIN;
  unsigned short* pWg1  = pWtr + (size_t)TK * PW_TR1;
  unsigned short* pWg2  = pWg1 + (size_t)TK * PW_TR1;

  int total = PW_LIN + TK * (2 * PW_TR1 + PW_G21);
  convert_weights<<<(total + 255) / 256, 256, 0, stream>>>(W_lin, W_tr, W_g1, W_g2, ws, total);

  int rowTiles = (N + 15) / 16;
  linear_kernel<<<(rowTiles + 3) / 4, 128, 0, stream>>>(h, pWlin, b_lin, (float*)d_out, N);

  size_t smemBytes = (size_t)LDS_TOTAL * sizeof(unsigned short);   // 151552 B
  edge_kernel<<<TK * NB, 128, smemBytes, stream>>>(h, pairs, degrees, scatter,
      pWtr, pWg1, pWg2, b_tr, b_g1, b_g2, eps, (float*)d_out, E);
}
